// MultiHeadSelfAttention_4260607558142
// MI455X (gfx1250) — matrix operand order repaired and checked by reading, not compiled, not run on hardware
//
#include <hip/hip_runtime.h>
#include <hip/hip_bf16.h>
#include <stdint.h>

// ---------------------------------------------------------------------------
// MultiHeadSelfAttention for MI455X (gfx1250, wave32, WMMA).
// All contractions on v_wmma_f32_16x16x32_bf16 (f32 accumulate).
// Global->LDS staging uses GLOBAL_LOAD_ASYNC_TO_LDS_B128 (ASYNCcnt) with
// single-barrier double buffering when the builtin exists; sync fallback else.
// ---------------------------------------------------------------------------

typedef __attribute__((ext_vector_type(16))) __bf16 v16bf;
typedef __attribute__((ext_vector_type(8)))  float  v8f;
typedef __attribute__((ext_vector_type(4)))  int    v4i;

union FragBF {          // 16 bf16 = 8 VGPRs = one WMMA A/B fragment (wave32)
  v16bf v;
  uint4 q[2];
};
static_assert(sizeof(FragBF) == 32, "frag size");

#define BATCH  2
#define SEQ    2048
#define DMODEL 1024
#define NH     16
#define DKH    64
#define MTOT   (BATCH * SEQ)   // 4096 rows

#define AS1 __attribute__((address_space(1)))
#define AS3 __attribute__((address_space(3)))

#if defined(__has_builtin)
#if __has_builtin(__builtin_amdgcn_global_load_async_to_lds_b128) && \
    __has_builtin(__builtin_amdgcn_s_wait_asynccnt)
#define USE_ASYNC_LDS 1
#endif
#endif
#ifndef USE_ASYNC_LDS
#define USE_ASYNC_LDS 0
#endif

// 16-byte global -> LDS copy (async DMA when available)
__device__ __forceinline__ void copy16_g2l(unsigned short* lds_dst,
                                           const unsigned short* gsrc) {
#if USE_ASYNC_LDS
  __builtin_amdgcn_global_load_async_to_lds_b128(
      (AS1 v4i*)gsrc, (AS3 v4i*)lds_dst, 0, 0);
#else
  *(uint4*)lds_dst = *(const uint4*)gsrc;
#endif
}
__device__ __forceinline__ void wait_stage() {
#if USE_ASYNC_LDS
  __builtin_amdgcn_s_wait_asynccnt(0);
#endif
}

__device__ __forceinline__ unsigned short f2bf(float f) {
  // round-to-nearest-even f32 -> bf16
  unsigned u = __float_as_uint(f);
  unsigned r = u + 0x7FFFu + ((u >> 16) & 1u);
  return (unsigned short)(r >> 16);
}

// ---------------------------------------------------------------------------
// f32 -> bf16 conversion, 4 elements/thread
// ---------------------------------------------------------------------------
__global__ __launch_bounds__(256) void convert_f32_bf16(
    const float* __restrict__ in, unsigned short* __restrict__ out, int n4) {
  int i = blockIdx.x * blockDim.x + threadIdx.x;
  int stride = gridDim.x * blockDim.x;
  for (; i < n4; i += stride) {
    float4 f = ((const float4*)in)[i];
    ushort4 o;
    o.x = f2bf(f.x); o.y = f2bf(f.y); o.z = f2bf(f.z); o.w = f2bf(f.w);
    ((ushort4*)out)[i] = o;
  }
}

// ---------------------------------------------------------------------------
// GEMM: C(M=4096, N=1024) = A(M,K=1024) * B(N,K)^T + bias
// 256 threads (8 waves), 128x64 C tile, K-step 64, double-buffered LDS,
// single barrier per K-step.  Each wave: 16(M) x 64(N) = 4 WMMA accumulators,
// 8 WMMAs per stage.
// MODE 0: write bf16 (B,H,S,64)   (Q, K)
// MODE 1: write bf16 (B,H,64,S)   (V transposed)
// MODE 2: write f32 row-major     (output projection -> d_out)
// ---------------------------------------------------------------------------
#define BM 128
#define BN 64
#define BK 64

template <int MODE>
__global__ __launch_bounds__(256) void gemm_bf16_wmma(
    const unsigned short* __restrict__ Ag,
    const unsigned short* __restrict__ Bg,
    const float* __restrict__ bias,
    unsigned short* __restrict__ outb,
    float* __restrict__ outf) {
  __shared__ __align__(16) unsigned short sA[2][BM * BK];  // 2 x 16 KB
  __shared__ __align__(16) unsigned short sB[2][BN * BK];  // 2 x  8 KB

  const int n0   = blockIdx.x * BN;
  const int m0   = blockIdx.y * BM;
  const int tid  = threadIdx.x;
  const int lane = tid & 31;
  const int wave = tid >> 5;
  const int hl   = lane >> 4;
  const int row  = lane & 15;
  const int kb   = hl * 8;    // A-frag K base (ISA 7.12.2)
  const int kh   = hl * 16;   // unused (B-frag also uses kb)

  auto stage = [&](int buf, int kk) {
#pragma unroll
    for (int i = 0; i < 4; ++i) {          // A tile: 1024 x 16B chunks
      int c = tid + 256 * i;
      int r = c >> 3, off = (c & 7) * 8;
      copy16_g2l(&sA[buf][r * BK + off],
                 &Ag[(size_t)(m0 + r) * DMODEL + kk + off]);
    }
#pragma unroll
    for (int i = 0; i < 2; ++i) {          // B tile: 512 x 16B chunks
      int c = tid + 256 * i;
      int r = c >> 3, off = (c & 7) * 8;
      copy16_g2l(&sB[buf][r * BK + off],
                 &Bg[(size_t)(n0 + r) * DMODEL + kk + off]);
    }
  };

  v8f acc[4];
#pragma unroll
  for (int j = 0; j < 4; ++j)
#pragma unroll
    for (int e = 0; e < 8; ++e) acc[j][e] = 0.0f;

  stage(0, 0);
  for (int it = 0; it < DMODEL / BK; ++it) {
    const int buf = it & 1;
    wait_stage();          // own async loads done
    __syncthreads();       // everyone's loads done; prev compute on buf^1 done
    if (it + 1 < DMODEL / BK) stage(buf ^ 1, (it + 1) * BK);

#pragma unroll
    for (int kc = 0; kc < 2; ++kc) {
      FragBF a;
      a.q[0] = *(const uint4*)&sA[buf][(wave * 16 + row) * BK + kc * 32 + kb];
      a.q[1] = *(const uint4*)&sA[buf][(wave * 16 + row) * BK + kc * 32 + kb + 16];
#pragma unroll
      for (int j = 0; j < 4; ++j) {
        FragBF bfr;
        bfr.q[0] = *(const uint4*)&sB[buf][(j * 16 + row) * BK + kc * 32 + kb];
        bfr.q[1] = *(const uint4*)&sB[buf][(j * 16 + row) * BK + kc * 32 + kb + 16];
        acc[j] = __builtin_amdgcn_wmma_f32_16x16x32_bf16(
            false, a.v, false, bfr.v, (short)0, acc[j], false, false);
      }
    }
  }

  // C layout: element e, lanes 0-15 -> M=e, lanes 16-31 -> M=e+8; N = lane&15
#pragma unroll
  for (int j = 0; j < 4; ++j) {
    const int n    = n0 + j * 16 + row;
    const float bv = bias[n];
#pragma unroll
    for (int e = 0; e < 8; ++e) {
      const int m   = m0 + wave * 16 + e + 8 * hl;
      const float v = acc[j][e] + bv;
      if (MODE == 2) {
        outf[(size_t)m * DMODEL + n] = v;
      } else {
        const int b = m >> 11, s = m & (SEQ - 1);
        const int h = n >> 6,  d = n & (DKH - 1);
        if (MODE == 0)
          outb[((size_t)(b * NH + h) * SEQ + s) * DKH + d] = f2bf(v);
        else
          outb[((size_t)(b * NH + h) * DKH + d) * SEQ + s] = f2bf(v);
      }
    }
  }
}

// ---------------------------------------------------------------------------
// Flash attention: grid (S/128, H, B), 256 threads (8 waves), wave = 16 rows.
// K tiles (64 keys x 64 d) and V^T tiles (64 d x 64 keys) double-buffered in
// LDS via async copies; online softmax in f32; P through per-wave LDS scratch.
// ---------------------------------------------------------------------------
__global__ __launch_bounds__(256) void attn_wmma(
    const unsigned short* __restrict__ Qg,   // (B,H,S,64) bf16
    const unsigned short* __restrict__ Kg,   // (B,H,S,64) bf16
    const unsigned short* __restrict__ VTg,  // (B,H,64,S) bf16
    const int* __restrict__ amask,           // (B,S)
    unsigned short* __restrict__ ctx) {      // (B,S,1024) bf16
  __shared__ __align__(16) unsigned short ldsK[2][64 * 64];  // 2 x 8 KB
  __shared__ __align__(16) unsigned short ldsV[2][64 * 64];  // 2 x 8 KB
  __shared__ __align__(16) unsigned short ldsP[8][16 * 64];  // 16 KB

  const int qt = blockIdx.x, h = blockIdx.y, b = blockIdx.z;
  const int tid  = threadIdx.x;
  const int lane = tid & 31;
  const int wave = tid >> 5;
  const int hl   = lane >> 4;
  const int row  = lane & 15;
  const int kb   = hl * 8;
  const int kh   = hl * 16;

  const size_t kbase = (size_t)(b * NH + h) * SEQ * DKH;
  const size_t vbase = (size_t)(b * NH + h) * DKH * SEQ;

  auto stage = [&](int buf, int kt) {
#pragma unroll
    for (int i = 0; i < 2; ++i) {          // 512 chunks each for K and V^T
      int c = tid + 256 * i;
      int r = c >> 3, off = (c & 7) * 8;
      copy16_g2l(&ldsK[buf][r * 64 + off],
                 &Kg[kbase + (size_t)(kt * 64 + r) * DKH + off]);
      copy16_g2l(&ldsV[buf][r * 64 + off],
                 &VTg[vbase + (size_t)r * SEQ + kt * 64 + off]);
    }
  };

  // preload the wave's Q fragments: 16 rows x 64 d = two 16x32 chunks
  FragBF qf[2];
  {
    const size_t qbase =
        ((size_t)(b * NH + h) * SEQ + qt * 128 + wave * 16 + row) * DKH;
#pragma unroll
    for (int c2 = 0; c2 < 2; ++c2) {
      qf[c2].q[0] = *(const uint4*)&Qg[qbase + c2 * 32 + kb];
      qf[c2].q[1] = *(const uint4*)&Qg[qbase + c2 * 32 + kb + 16];
    }
  }

  v8f o[4];
  float m_run[8], l_run[8];
#pragma unroll
  for (int j = 0; j < 4; ++j)
#pragma unroll
    for (int e = 0; e < 8; ++e) o[j][e] = 0.0f;
#pragma unroll
  for (int e = 0; e < 8; ++e) { m_run[e] = -3.0e38f; l_run[e] = 0.0f; }

  stage(0, 0);
  for (int kt = 0; kt < SEQ / 64; ++kt) {
    const int buf = kt & 1;
    wait_stage();
    __syncthreads();     // tile(buf) complete; all waves done with tile(buf^1)
    if (kt + 1 < SEQ / 64) stage(buf ^ 1, kt + 1);

    // scores S = Q * K^T  (contraction over d = 2 chunks of 32)
    v8f sc[4];
#pragma unroll
    for (int j = 0; j < 4; ++j)
#pragma unroll
      for (int e = 0; e < 8; ++e) sc[j][e] = 0.0f;
#pragma unroll
    for (int j = 0; j < 4; ++j) {
#pragma unroll
      for (int c2 = 0; c2 < 2; ++c2) {
        FragBF bf;  // B = [d, key]: lane = key column, d contiguous
        bf.q[0] = *(const uint4*)&ldsK[buf][(j * 16 + row) * 64 + c2 * 32 + kb];
        bf.q[1] = *(const uint4*)&ldsK[buf][(j * 16 + row) * 64 + c2 * 32 + kb + 16];
        sc[j] = __builtin_amdgcn_wmma_f32_16x16x32_bf16(
            false, qf[c2].v, false, bf.v, (short)0, sc[j], false, false);
      }
    }

    // scale 1/sqrt(64) and mask
#pragma unroll
    for (int j = 0; j < 4; ++j) {
      const int kcol  = kt * 64 + j * 16 + row;
      const float mng = (amask[b * SEQ + kcol] == 0) ? -1.0e9f : 0.0f;
#pragma unroll
      for (int e = 0; e < 8; ++e) sc[j][e] = sc[j][e] * 0.125f + mng;
    }

    // online softmax: row r = e + 8*hl lives across the 16-lane half
    float nm[8], alpha[8], rs[8];
#pragma unroll
    for (int e = 0; e < 8; ++e) {
      float v = fmaxf(fmaxf(sc[0][e], sc[1][e]), fmaxf(sc[2][e], sc[3][e]));
      v = fmaxf(v, __shfl_xor(v, 1, 32));
      v = fmaxf(v, __shfl_xor(v, 2, 32));
      v = fmaxf(v, __shfl_xor(v, 4, 32));
      v = fmaxf(v, __shfl_xor(v, 8, 32));
      nm[e]    = fmaxf(m_run[e], v);
      alpha[e] = __expf(m_run[e] - nm[e]);
      m_run[e] = nm[e];
      rs[e]    = 0.0f;
    }
#pragma unroll
    for (int j = 0; j < 4; ++j) {
#pragma unroll
      for (int e = 0; e < 8; ++e) {
        const float p = __expf(sc[j][e] - nm[e]);
        rs[e] += p;
        ldsP[wave][(e + 8 * hl) * 64 + j * 16 + row] = f2bf(p);
      }
    }
#pragma unroll
    for (int e = 0; e < 8; ++e) {
      float r = rs[e];
      r += __shfl_xor(r, 1, 32);
      r += __shfl_xor(r, 2, 32);
      r += __shfl_xor(r, 4, 32);
      r += __shfl_xor(r, 8, 32);
      l_run[e] = l_run[e] * alpha[e] + r;
    }
    __syncthreads();     // P rows (cross-lane through LDS) visible

    // rescale running context, then ctx += P * V (contraction over 64 keys)
#pragma unroll
    for (int j = 0; j < 4; ++j)
#pragma unroll
      for (int e = 0; e < 8; ++e) o[j][e] *= alpha[e];

#pragma unroll
    for (int c2 = 0; c2 < 2; ++c2) {
      FragBF pa;  // A = P row-major
      pa.q[0] = *(const uint4*)&ldsP[wave][row * 64 + c2 * 32 + kb];
      pa.q[1] = *(const uint4*)&ldsP[wave][row * 64 + c2 * 32 + kb + 16];
#pragma unroll
      for (int j = 0; j < 4; ++j) {
        FragBF bf;  // B = [key, d]: lane = d column, key contiguous from V^T
        bf.q[0] = *(const uint4*)&ldsV[buf][(j * 16 + row) * 64 + c2 * 32 + kb];
        bf.q[1] = *(const uint4*)&ldsV[buf][(j * 16 + row) * 64 + c2 * 32 + kb + 16];
        o[j] = __builtin_amdgcn_wmma_f32_16x16x32_bf16(
            false, pa.v, false, bf.v, (short)0, o[j], false, false);
      }
    }
  }

  // finalize: divide by l, store ctx (B,S,H*64) bf16
  float inv[8];
#pragma unroll
  for (int e = 0; e < 8; ++e) inv[e] = 1.0f / l_run[e];
#pragma unroll
  for (int j = 0; j < 4; ++j)
#pragma unroll
    for (int e = 0; e < 8; ++e) {
      const int q = qt * 128 + wave * 16 + e + 8 * hl;
      const int d = h * DKH + j * 16 + row;
      ctx[((size_t)b * SEQ + q) * DMODEL + d] = f2bf(o[j][e] * inv[e]);
    }
}

// ---------------------------------------------------------------------------
// Launch
// ---------------------------------------------------------------------------
extern "C" void kernel_launch(void* const* d_in, const int* in_sizes, int n_in,
                              void* d_out, int out_size, void* d_ws, size_t ws_size,
                              hipStream_t stream) {
  const float* x   = (const float*)d_in[0];
  const int*   am  = (const int*)d_in[1];
  const float* w_q = (const float*)d_in[2];
  const float* b_q = (const float*)d_in[3];
  const float* w_k = (const float*)d_in[4];
  const float* b_k = (const float*)d_in[5];
  const float* w_v = (const float*)d_in[6];
  const float* b_v = (const float*)d_in[7];
  const float* w_o = (const float*)d_in[8];
  const float* b_o = (const float*)d_in[9];
  float* out = (float*)d_out;

  // workspace layout (48 MB total), every region fully written before read
  char* ws = (char*)d_ws;
  unsigned short* xb  = (unsigned short*)(ws + ( 0ull << 20));  // 8 MB
  unsigned short* wqb = (unsigned short*)(ws + ( 8ull << 20));  // 2 MB
  unsigned short* wkb = (unsigned short*)(ws + (10ull << 20));  // 2 MB
  unsigned short* wvb = (unsigned short*)(ws + (12ull << 20));  // 2 MB
  unsigned short* wob = (unsigned short*)(ws + (14ull << 20));  // 2 MB
  unsigned short* Qb  = (unsigned short*)(ws + (16ull << 20));  // 8 MB
  unsigned short* Kb  = (unsigned short*)(ws + (24ull << 20));  // 8 MB
  unsigned short* VTb = (unsigned short*)(ws + (32ull << 20));  // 8 MB
  unsigned short* Cb  = (unsigned short*)(ws + (40ull << 20));  // 8 MB

  convert_f32_bf16<<<512, 256, 0, stream>>>(x,   xb,  MTOT * DMODEL / 4);
  convert_f32_bf16<<<256, 256, 0, stream>>>(w_q, wqb, DMODEL * DMODEL / 4);
  convert_f32_bf16<<<256, 256, 0, stream>>>(w_k, wkb, DMODEL * DMODEL / 4);
  convert_f32_bf16<<<256, 256, 0, stream>>>(w_v, wvb, DMODEL * DMODEL / 4);
  convert_f32_bf16<<<256, 256, 0, stream>>>(w_o, wob, DMODEL * DMODEL / 4);

  dim3 gg(DMODEL / BN, MTOT / BM);   // (16, 32)
  gemm_bf16_wmma<0><<<gg, 256, 0, stream>>>(xb, wqb, b_q, Qb,  nullptr);
  gemm_bf16_wmma<0><<<gg, 256, 0, stream>>>(xb, wkb, b_k, Kb,  nullptr);
  gemm_bf16_wmma<1><<<gg, 256, 0, stream>>>(xb, wvb, b_v, VTb, nullptr);

  attn_wmma<<<dim3(SEQ / 128, NH, BATCH), 256, 0, stream>>>(Qb, Kb, VTb, am, Cb);

  gemm_bf16_wmma<2><<<gg, 256, 0, stream>>>(Cb, wob, b_o, nullptr, out);
}
